// OmegaMoELayer_26130581029532
// MI455X (gfx1250) — 8-bit matrix operand order repaired and checked by reading, compiled, not run on hardware
//
#include <hip/hip_runtime.h>
#include <hip/hip_bf16.h>


typedef int v8i __attribute__((ext_vector_type(8)));

#define T_TOK 16384
#define H_DIM 2048
#define D_DIM 2048
#define E_NUM 8
#define DH    (D_DIM * H_DIM)
#define EPSQ  1e-5f


__device__ __forceinline__ v8i load_fragA_off(const signed char* __restrict__ base, unsigned off) {
    int2 q0 = *(const int2*)(base + off);
    int2 q1 = *(const int2*)(base + off + 16);
    int2 q2 = *(const int2*)(base + off + 32);
    int2 q3 = *(const int2*)(base + off + 48);
    v8i a = {q0.x, q0.y, q1.x, q1.y, q2.x, q2.y, q3.x, q3.y};
    return a;
}

__device__ __forceinline__ v8i load_fragB_off(const signed char* __restrict__ base, unsigned off) {
    int2 q0 = *(const int2*)(base + off);
    int2 q1 = *(const int2*)(base + off + 16);
    int2 q2 = *(const int2*)(base + off + 32);
    int2 q3 = *(const int2*)(base + off + 48);
    v8i b = {q0.x, q0.y, q1.x, q1.y, q2.x, q2.y, q3.x, q3.y};
    return b;
}

__device__ __forceinline__ int quant_clamp(float x, float lo, float hi) {
    float q = rintf(x);
    q = fminf(fmaxf(q, lo), hi);
    return (int)q;
}


__global__ void init_kernel(int* __restrict__ cnt) {
    if (threadIdx.x < E_NUM) cnt[threadIdx.x] = 0;
}


__global__ void router_quant_kernel(const float* __restrict__ x,
                                    const float* __restrict__ rw,
                                    signed char* __restrict__ xq,
                                    float* __restrict__ xsc,
                                    int* __restrict__ idx,
                                    int* __restrict__ cnt,
                                    int* __restrict__ list) {
    __shared__ float xs[H_DIM];
    __shared__ float red[256];
    __shared__ float lg[256 * 8];
    __shared__ float s_s127;

    const int t   = blockIdx.x;
    const int tid = threadIdx.x;

    float am = 0.f;
    for (int k = tid; k < H_DIM; k += 256) {
        float v = x[(size_t)t * H_DIM + k];
        xs[k] = v;
        am = fmaxf(am, fabsf(v));
    }
    red[tid] = am;
    __syncthreads();
    for (int s = 128; s > 0; s >>= 1) {
        if (tid < s) red[tid] = fmaxf(red[tid], red[tid + s]);
        __syncthreads();
    }
    if (tid == 0) {
        float amax = fmaxf(red[0], EPSQ);
        s_s127 = 127.f / amax;
        xsc[t] = amax / 127.f;
    }

    float acc[E_NUM];
#pragma unroll
    for (int e = 0; e < E_NUM; ++e) acc[e] = 0.f;
    for (int k = tid; k < H_DIM; k += 256) {
        float v = xs[k];
#pragma unroll
        for (int e = 0; e < E_NUM; ++e)
            acc[e] = fmaf(v, rw[(size_t)e * H_DIM + k], acc[e]);
    }
#pragma unroll
    for (int e = 0; e < E_NUM; ++e) lg[tid * 8 + e] = acc[e];
    __syncthreads();
    for (int s = 128; s > 0; s >>= 1) {
        if (tid < s) {
#pragma unroll
            for (int e = 0; e < E_NUM; ++e)
                lg[tid * 8 + e] += lg[(tid + s) * 8 + e];
        }
        __syncthreads();
    }
    if (tid == 0) {
        int best = 0; float bv = lg[0];
#pragma unroll
        for (int e = 1; e < E_NUM; ++e)
            if (lg[e] > bv) { bv = lg[e]; best = e; }
        idx[t] = best;
        int pos = atomicAdd(&cnt[best], 1);
        list[best * T_TOK + pos] = t;
    }
    __syncthreads();

    float s127 = s_s127;
    for (int k = tid; k < H_DIM; k += 256)
        xq[(size_t)t * H_DIM + k] = (signed char)quant_clamp(xs[k] * s127, -128.f, 127.f);
}


__global__ void wsum_kernel(const float* __restrict__ wg,
                            const float* __restrict__ wu,
                            const float* __restrict__ wd,
                            float* __restrict__ wpart) {
    const int j = blockIdx.y;
    const float* src = (j < 8)  ? wg + (size_t)j * DH
                     : (j < 16) ? wu + (size_t)(j - 8) * DH
                                : wd + (size_t)(j - 16) * DH;
    float sum = 0.f;
    for (size_t i = (size_t)blockIdx.x * 256 + threadIdx.x; i < (size_t)DH; i += 64 * 256)
        sum += fabsf(src[i]);
    __shared__ float red[256];
    red[threadIdx.x] = sum;
    __syncthreads();
    for (int s = 128; s > 0; s >>= 1) {
        if (threadIdx.x < s) red[threadIdx.x] += red[threadIdx.x + s];
        __syncthreads();
    }
    if (threadIdx.x == 0) wpart[j * 64 + blockIdx.x] = red[0];
}

__global__ void wscale_kernel(const float* __restrict__ wpart, float* __restrict__ wsc) {
    int j = threadIdx.x;
    if (j < 24) {
        float s = 0.f;
        for (int i = 0; i < 64; ++i) s += wpart[j * 64 + i];
        float mean = s / (float)DH;
        wsc[j] = fmaxf(mean, EPSQ);
    }
}


__global__ void wquant_kernel(const float* __restrict__ wg,
                              const float* __restrict__ wu,
                              const float* __restrict__ wd,
                              const float* __restrict__ wsc,
                              signed char* __restrict__ wq_all) {
    size_t i = ((size_t)blockIdx.x * 256 + threadIdx.x) * 4;
    if (i >= 3ull * E_NUM * DH) return;
    int j = (int)(i >> 22);
    const float* src; size_t off;
    if (j < 8)       { src = wg; off = i; }
    else if (j < 16) { src = wu; off = i - 8ull * DH; }
    else             { src = wd; off = i - 16ull * DH; }
    float4 v = *(const float4*)(src + off);
    float s = 1.f / wsc[j];
    int q0 = quant_clamp(v.x * s, -1.f, 1.f);
    int q1 = quant_clamp(v.y * s, -1.f, 1.f);
    int q2 = quant_clamp(v.z * s, -1.f, 1.f);
    int q3 = quant_clamp(v.w * s, -1.f, 1.f);
    int packed = (q0 & 255) | ((q1 & 255) << 8) | ((q2 & 255) << 16) | (q3 << 24);
    *(int*)(wq_all + i) = packed;
}


__global__ void __launch_bounds__(128, 1)
gemm1_kernel(const signed char* __restrict__ xq,
             const signed char* __restrict__ wgq,
             const signed char* __restrict__ wuq,
             const float* __restrict__ xsc,
             const float* __restrict__ wsc,
             const int* __restrict__ cnt,
             const int* __restrict__ list,
             float* __restrict__ h) {
    const int e     = blockIdx.x;
    const int tile  = blockIdx.y;
    const int cnt_e = cnt[e];
    const int rowbase = tile * 32;
    if (rowbase >= cnt_e) return;

    const int lane = threadIdx.x & 31;
    const int wave = threadIdx.x >> 5;
    const int half = lane >> 4;
    const int nlo  = lane & 15;
    const int colbase = blockIdx.z * 256 + wave * 64;
    const int* lst = list + e * T_TOK;

    const int ts0 = rowbase + nlo;
    const int ts1 = rowbase + 16 + nlo;
    const int tok0 = lst[ts0 < cnt_e ? ts0 : rowbase];
    const int tok1 = lst[ts1 < cnt_e ? ts1 : rowbase];

    const signed char* wgq_e = wgq + (size_t)e * DH;
    const signed char* wuq_e = wuq + (size_t)e * DH;
    unsigned offA0 = (unsigned)tok0 * H_DIM + half * 8;
    unsigned offA1 = (unsigned)tok1 * H_DIM + half * 8;
    unsigned offB  = (unsigned)(colbase + nlo) * H_DIM + half * 8;

    const v8i zero = {0, 0, 0, 0, 0, 0, 0, 0};
    v8i accg[2][4], accu[2][4];
#pragma unroll
    for (int t = 0; t < 4; ++t) {
        accg[0][t] = zero; accg[1][t] = zero;
        accu[0][t] = zero; accu[1][t] = zero;
    }

    for (int kc = 0; kc < H_DIM; kc += 64) {
        v8i a0 = load_fragA_off(xq, offA0);
        v8i a1 = load_fragA_off(xq, offA1);
#pragma unroll
        for (int t = 0; t < 4; ++t) {
            v8i b = load_fragB_off(wgq_e, offB + (unsigned)(t * 16 * H_DIM));
            accg[0][t] = __builtin_amdgcn_wmma_i32_16x16x64_iu8(true, a0, true, b, accg[0][t], false, false);
            accg[1][t] = __builtin_amdgcn_wmma_i32_16x16x64_iu8(true, a1, true, b, accg[1][t], false, false);
        }
#pragma unroll
        for (int t = 0; t < 4; ++t) {
            v8i b = load_fragB_off(wuq_e, offB + (unsigned)(t * 16 * H_DIM));
            accu[0][t] = __builtin_amdgcn_wmma_i32_16x16x64_iu8(true, a0, true, b, accu[0][t], false, false);
            accu[1][t] = __builtin_amdgcn_wmma_i32_16x16x64_iu8(true, a1, true, b, accu[1][t], false, false);
        }
        offA0 += 64; offA1 += 64; offB += 64;
    }

    const float wgd = wsc[e];
    const float wud = wsc[8 + e];
#pragma unroll
    for (int s = 0; s < 2; ++s) {
#pragma unroll
        for (int r = 0; r < 8; ++r) {
            int ts = rowbase + s * 16 + r + 8 * half;
            bool valid = ts < cnt_e;
            int tok = lst[valid ? ts : rowbase];
            float xd = xsc[tok];
#pragma unroll
            for (int t = 0; t < 4; ++t) {
                int col  = colbase + t * 16 + nlo;
                float gf = (float)accg[s][t][r] * xd * wgd;
                float uf = (float)accu[s][t][r] * xd * wud;
                float rl = gf > 0.f ? gf : 0.f;
                if (valid) h[(size_t)tok * D_DIM + col] = rl * rl * uf;
            }
        }
    }
}


__global__ void hquant_kernel(const float* __restrict__ h,
                              signed char* __restrict__ hq,
                              float* __restrict__ hsc) {
    const int t   = blockIdx.x;
    const int tid = threadIdx.x;
    __shared__ float red[256];
    float am = 0.f;
    for (int k = tid; k < D_DIM; k += 256)
        am = fmaxf(am, fabsf(h[(size_t)t * D_DIM + k]));
    red[tid] = am;
    __syncthreads();
    for (int s = 128; s > 0; s >>= 1) {
        if (tid < s) red[tid] = fmaxf(red[tid], red[tid + s]);
        __syncthreads();
    }
    float amax = fmaxf(red[0], EPSQ);
    float s127 = 127.f / amax;
    if (tid == 0) hsc[t] = amax / 127.f;
    for (int k = tid; k < D_DIM; k += 256)
        hq[(size_t)t * D_DIM + k] =
            (signed char)quant_clamp(h[(size_t)t * D_DIM + k] * s127, -128.f, 127.f);
}


__global__ void __launch_bounds__(128, 1)
gemm2_kernel(const signed char* __restrict__ hq,
             const signed char* __restrict__ wdq,
             const float* __restrict__ hsc,
             const float* __restrict__ wsc,
             const int* __restrict__ cnt,
             const int* __restrict__ list,
             float* __restrict__ out) {
    const int e     = blockIdx.x;
    const int tile  = blockIdx.y;
    const int cnt_e = cnt[e];
    const int rowbase = tile * 32;
    if (rowbase >= cnt_e) return;

    const int lane = threadIdx.x & 31;
    const int wave = threadIdx.x >> 5;
    const int half = lane >> 4;
    const int nlo  = lane & 15;
    const int colbase = blockIdx.z * 256 + wave * 64;
    const int* lst = list + e * T_TOK;

    const int ts0 = rowbase + nlo;
    const int ts1 = rowbase + 16 + nlo;
    const int tok0 = lst[ts0 < cnt_e ? ts0 : rowbase];
    const int tok1 = lst[ts1 < cnt_e ? ts1 : rowbase];

    const signed char* wdq_e = wdq + (size_t)e * DH;
    unsigned offA0 = (unsigned)tok0 * D_DIM + half * 8;
    unsigned offA1 = (unsigned)tok1 * D_DIM + half * 8;
    unsigned offB  = (unsigned)(colbase + nlo) * D_DIM + half * 8;

    const v8i zero = {0, 0, 0, 0, 0, 0, 0, 0};
    v8i acc[2][4];
#pragma unroll
    for (int t = 0; t < 4; ++t) { acc[0][t] = zero; acc[1][t] = zero; }

    for (int kc = 0; kc < D_DIM; kc += 64) {
        v8i a0 = load_fragA_off(hq, offA0);
        v8i a1 = load_fragA_off(hq, offA1);
#pragma unroll
        for (int t = 0; t < 4; ++t) {
            v8i b = load_fragB_off(wdq_e, offB + (unsigned)(t * 16 * D_DIM));
            acc[0][t] = __builtin_amdgcn_wmma_i32_16x16x64_iu8(true, a0, true, b, acc[0][t], false, false);
            acc[1][t] = __builtin_amdgcn_wmma_i32_16x16x64_iu8(true, a1, true, b, acc[1][t], false, false);
        }
        offA0 += 64; offA1 += 64; offB += 64;
    }

    const float wdd = wsc[16 + e];
#pragma unroll
    for (int s = 0; s < 2; ++s) {
#pragma unroll
        for (int r = 0; r < 8; ++r) {
            int ts = rowbase + s * 16 + r + 8 * half;
            bool valid = ts < cnt_e;
            int tok = lst[valid ? ts : rowbase];
            float hd = hsc[tok];
#pragma unroll
            for (int t = 0; t < 4; ++t) {
                int col = colbase + t * 16 + nlo;
                if (valid) out[(size_t)tok * H_DIM + col] = (float)acc[s][t][r] * hd * wdd;
            }
        }
    }
}


extern "C" void kernel_launch(void* const* d_in, const int* in_sizes, int n_in,
                              void* d_out, int out_size, void* d_ws, size_t ws_size,
                              hipStream_t stream) {
    const float* x  = (const float*)d_in[0];
    const float* rw = (const float*)d_in[1];
    const float* wg = (const float*)d_in[2];
    const float* wu = (const float*)d_in[3];
    const float* wd = (const float*)d_in[4];
    float* out = (float*)d_out;

    constexpr size_t o_xq    = 0;
    constexpr size_t o_wgq   = o_xq    + (size_t)T_TOK * H_DIM;
    constexpr size_t o_wuq   = o_wgq   + (size_t)E_NUM * DH;
    constexpr size_t o_wdq   = o_wuq   + (size_t)E_NUM * DH;
    constexpr size_t o_hq    = o_wdq   + (size_t)E_NUM * DH;
    constexpr size_t o_h     = o_hq    + (size_t)T_TOK * D_DIM;
    constexpr size_t o_xs    = o_h     + (size_t)T_TOK * D_DIM * 4;
    constexpr size_t o_hs    = o_xs    + (size_t)T_TOK * 4;
    constexpr size_t o_idx   = o_hs    + (size_t)T_TOK * 4;
    constexpr size_t o_cnt   = o_idx   + (size_t)T_TOK * 4;
    constexpr size_t o_list  = o_cnt   + 64;
    constexpr size_t o_wpart = o_list  + (size_t)E_NUM * T_TOK * 4;
    constexpr size_t o_wsc   = o_wpart + 24 * 64 * 4;

    char* ws = (char*)d_ws;
    signed char* xq  = (signed char*)(ws + o_xq);
    signed char* wgq = (signed char*)(ws + o_wgq);
    signed char* wuq = (signed char*)(ws + o_wuq);
    signed char* wdq = (signed char*)(ws + o_wdq);
    signed char* hq  = (signed char*)(ws + o_hq);
    float* h     = (float*)(ws + o_h);
    float* xsc   = (float*)(ws + o_xs);
    float* hsc   = (float*)(ws + o_hs);
    int*   idx   = (int*)(ws + o_idx);
    int*   cnt   = (int*)(ws + o_cnt);
    int*   list  = (int*)(ws + o_list);
    float* wpart = (float*)(ws + o_wpart);
    float* wsc   = (float*)(ws + o_wsc);

    init_kernel<<<1, 32, 0, stream>>>(cnt);

    router_quant_kernel<<<T_TOK, 256, 0, stream>>>(x, rw, xq, xsc, idx, cnt, list);

    wsum_kernel<<<dim3(64, 24), 256, 0, stream>>>(wg, wu, wd, wpart);
    wscale_kernel<<<1, 32, 0, stream>>>(wpart, wsc);
    {
        size_t n4 = 3ull * E_NUM * DH / 4;
        wquant_kernel<<<(unsigned)((n4 + 255) / 256), 256, 0, stream>>>(wg, wu, wd, wsc, wgq);
    }

    gemm1_kernel<<<dim3(E_NUM, T_TOK / 32, D_DIM / 256), 128, 0, stream>>>(
        xq, wgq, wuq, xsc, wsc, cnt, list, h);

    hquant_kernel<<<T_TOK, 256, 0, stream>>>(h, hq, hsc);

    gemm2_kernel<<<dim3(E_NUM, T_TOK / 32, H_DIM / 256), 128, 0, stream>>>(
        hq, wdq, hsc, wsc, cnt, list, out);
}
